// BrosEncoder_64544768524554
// MI455X (gfx1250) — matrix operand order repaired and checked by reading, not compiled, not run on hardware
//
#include <hip/hip_runtime.h>

// ---------------------------------------------------------------------------
// BROS encoder for MI455X (gfx1250): bf16 WMMA everywhere, fp32 accumulate.
// GEMM tiles staged to LDS with gfx1250 async global->LDS transfers
// (double-buffered, ASYNCcnt-synchronized) when the toolchain exposes them.
// ---------------------------------------------------------------------------

typedef __bf16 bf16_t;
typedef __attribute__((ext_vector_type(16))) __bf16 v16bf;
typedef __attribute__((ext_vector_type(8)))  __bf16 bf16x8;
typedef __attribute__((ext_vector_type(4)))  __bf16 bf16x4;
typedef __attribute__((ext_vector_type(8)))  float  v8f;
typedef __attribute__((ext_vector_type(4)))  int    v4i;

constexpr int Lc  = 12;
constexpr int Hc  = 768;
constexpr int NHc = 12;
constexpr int DHc = 64;
constexpr int Ic  = 3072;
constexpr int Bc  = 4;
constexpr int Sc  = 512;
constexpr int Mrows = Bc * Sc;           // 2048
constexpr float SCALEc = 0.125f;         // 1/sqrt(64)

// ---- async global->LDS support (gfx1250) ----------------------------------
#if defined(__HIP_DEVICE_COMPILE__) && defined(__has_builtin)
#if __has_builtin(__builtin_amdgcn_global_load_async_to_lds_b128)
#define ASYNC_LDS 1
#endif
#endif

#ifdef ASYNC_LDS
typedef __attribute__((address_space(1))) void gas_void;
typedef __attribute__((address_space(3))) void las_void;
typedef __attribute__((address_space(1))) v4i  gas_v4i;
typedef __attribute__((address_space(3))) v4i  las_v4i;
#endif

__device__ __forceinline__ void copy16(const bf16_t* g, bf16_t* l) {
#ifdef ASYNC_LDS
    __builtin_amdgcn_global_load_async_to_lds_b128(
        (gas_v4i*)(gas_void*)g,
        (las_v4i*)(las_void*)l, 0, 0);
#else
    *(bf16x8*)l = *(const bf16x8*)g;
#endif
}

__device__ __forceinline__ void async_wait() {
#ifdef ASYNC_LDS
#if __has_builtin(__builtin_amdgcn_s_wait_asynccnt)
    __builtin_amdgcn_s_wait_asynccnt(0);
#else
    asm volatile("s_wait_asynccnt 0" ::: "memory");
#endif
#endif
}

// ---------------------------------------------------------------------------
// Elementwise f32 -> bf16
// ---------------------------------------------------------------------------
__global__ void cvt_kernel(const float* __restrict__ in, bf16_t* __restrict__ out, long n) {
    long i = (long)blockIdx.x * blockDim.x + threadIdx.x;
    if (i < n) out[i] = (bf16_t)in[i];
}

// bbox [S,S,B,DH] f32 -> [B,S,S,DH] bf16
__global__ void bbox_tr_kernel(const float* __restrict__ in, bf16_t* __restrict__ out) {
    long idx = (long)blockIdx.x * blockDim.x + threadIdx.x;
    const long total = (long)Sc * Sc * Bc * DHc;
    if (idx >= total) return;
    int d = (int)(idx & (DHc - 1));
    long r = idx >> 6;                 // / DHc
    int b = (int)(r & 3); r >>= 2;     // / B
    int j = (int)(r & (Sc - 1));
    int i = (int)(r >> 9);             // / S
    out[((((long)b * Sc + i) * Sc + j) << 6) + d] = (bf16_t)in[idx];
}

// ---------------------------------------------------------------------------
// Generic batched bf16 WMMA GEMM:  C[z] = A[z] (MxK) * B[z] (KxN) (+bias)
// Block tile 128x128, K-step 32, 8 waves (2x4), each wave 64x32 = 8 WMMA accs.
// Double-buffered LDS; A tile (and B tile when stored [N,K]) staged with
// async global->LDS b128 transfers, pipelined one K-step ahead.
// BNK:   B stored as [N,K] rows (e.g. k-matrix for Q*K^T);  else [K,N] rows.
// ACCUM: C_f32 += acc (content scores accumulate onto bbox bias).
// GELU:  exact gelu epilogue.  OF32/OBF: which outputs to write.
// Batch offset for z: (z/zdiv)*s?1 + (z%zdiv)*s?2  (zdiv=1 -> linear).
// ---------------------------------------------------------------------------
constexpr int BM = 128, BN = 128, BK = 32;
constexpr int LDT = BK + 8;   // 40 elems = 80B row stride (16B aligned)

template<bool BNK, bool ACCUM, bool GELU, bool OF32, bool OBF>
__global__ __launch_bounds__(256) void gemm_kernel(
    const bf16_t* __restrict__ A, const bf16_t* __restrict__ Bm,
    const float* __restrict__ bias,
    float* __restrict__ cF, bf16_t* __restrict__ cB,
    int M, int N, int K, int lda, int ldb, int ldc,
    int zdiv, long sA1, long sA2, long sB1, long sB2, long sC1, long sC2)
{
    __shared__ bf16_t As[2][BM][LDT];
    __shared__ bf16_t Bs[2][BN][LDT];

    const int z = blockIdx.z;
    A  += (long)(z / zdiv) * sA1 + (long)(z % zdiv) * sA2;
    Bm += (long)(z / zdiv) * sB1 + (long)(z % zdiv) * sB2;
    const long coff = (long)(z / zdiv) * sC1 + (long)(z % zdiv) * sC2;
    if (cF) cF += coff;
    if (cB) cB += coff;

    const int mBase = blockIdx.y * BM;
    const int nBase = blockIdx.x * BN;
    const int tid  = threadIdx.x;
    const int lane = tid & 31;
    const int wid  = tid >> 5;
    const int mw = (wid >> 2) * 64;     // wave M offset within tile
    const int nw = (wid & 3) * 32;      // wave N offset within tile
    const int ml = lane & 15;
    const int hi = lane >> 4;

    v8f acc[4][2] = {};

    // stage A tile [128 rows][32 k] as 16B chunks (async when available)
    auto stageA = [&](int kk, int buf) {
        for (int c = tid; c < BM * (BK / 8); c += 256) {
            int m  = c >> 2;
            int kc = (c & 3) << 3;
            int gm = mBase + m;
            if (gm < M) copy16(A + (long)gm * lda + kk + kc, &As[buf][m][kc]);
        }
    };
    // stage B tile as [n][k]
    auto stageB = [&](int kk, int buf) {
        if constexpr (BNK) {
            for (int c = tid; c < BN * (BK / 8); c += 256) {
                int n  = c >> 2;
                int kc = (c & 3) << 3;
                int gn = nBase + n;
                if (gn < N) copy16(Bm + (long)gn * ldb + kk + kc, &Bs[buf][n][kc]);
            }
        } else {
            // transpose [K,N] -> LDS [n][k] (scatter; cannot use block async)
            for (int idx = tid; idx < BN * BK / 4; idx += 256) {
                int k  = idx >> 5;          // 0..31
                int nc = (idx & 31) << 2;   // 0..124
                int gn = nBase + nc;
                bf16x4 val = {};
                if (gn < N) val = *(const bf16x4*)(Bm + (long)(kk + k) * ldb + gn);
                Bs[buf][nc + 0][k] = val[0];
                Bs[buf][nc + 1][k] = val[1];
                Bs[buf][nc + 2][k] = val[2];
                Bs[buf][nc + 3][k] = val[3];
            }
        }
    };

    stageA(0, 0);
    stageB(0, 0);

    const int nk = K / BK;
    for (int t = 0; t < nk; t++) {
        const int cur = t & 1;
        async_wait();          // own async transfers for buffer `cur` complete
        __syncthreads();       // everyone's tile data visible; prev compute done

        if (t + 1 < nk) {      // pipeline: fill the other buffer
            stageA((t + 1) * BK, cur ^ 1);
            stageB((t + 1) * BK, cur ^ 1);
        }
        if (t + 2 < nk) {      // L2 prefetch two panels ahead
            int pm = mBase + (tid >> 1);
            if (pm < M) __builtin_prefetch(A + (long)pm * lda + (t + 2) * BK, 0, 1);
        }

        // ---- fragments + WMMA on buffer `cur` ----
        v16bf fa[4], fb[2];
        for (int mt = 0; mt < 4; mt++) {
            int row = mw + mt * 16 + ml;
            union { v16bf v; bf16x8 h[2]; } u;
            u.h[0] = *(const bf16x8*)&As[cur][row][hi * 8];        // K = hi*8 .. +7
            u.h[1] = *(const bf16x8*)&As[cur][row][16 + hi * 8];   // K = 16+hi*8 .. +7
            fa[mt] = u.v;
        }
        for (int nt = 0; nt < 2; nt++) {
            int col = nw + nt * 16 + ml;
            union { v16bf v; bf16x8 h[2]; } u;
            u.h[0] = *(const bf16x8*)&Bs[cur][col][hi * 8];        // K = hi*8 .. +7
            u.h[1] = *(const bf16x8*)&Bs[cur][col][16 + hi * 8];   // K = 16+hi*8 .. +7
            fb[nt] = u.v;
        }
        for (int mt = 0; mt < 4; mt++)
            for (int nt = 0; nt < 2; nt++)
                acc[mt][nt] = __builtin_amdgcn_wmma_f32_16x16x32_bf16(
                    false, fa[mt], false, fb[nt], (short)0, acc[mt][nt], false, false);
    }

    // ---- epilogue ----
    for (int mt = 0; mt < 4; mt++) {
        for (int nt = 0; nt < 2; nt++) {
            int n = nBase + nw + nt * 16 + ml;
            if (n >= N) continue;
            float bv = bias ? bias[n] : 0.0f;
            for (int r = 0; r < 8; r++) {
                int m = mBase + mw + mt * 16 + r + hi * 8;
                if (m >= M) continue;
                float v = acc[mt][nt][r] + bv;
                long off = (long)m * ldc + n;
                if constexpr (ACCUM) v += cF[off];
                if constexpr (GELU)  v = 0.5f * v * (1.0f + erff(v * 0.70710678118f));
                if constexpr (OF32)  cF[off] = v;
                if constexpr (OBF)   cB[off] = (bf16_t)v;
            }
        }
    }
}

// ---------------------------------------------------------------------------
// bbox positional bias: scores[b,h,i,j] = sum_d q[b,i,h,d] * bbox[b,i,j,d]
// For fixed (b,i) this is a GEMM: q_i [12(heads,pad16) x 64] * bbox[b,i]^T [64 x 512].
// One wave per (b,i); 4 waves per block. Writes the full scores tensor (the
// content-score GEMM then accumulates on top).
// ---------------------------------------------------------------------------
__global__ __launch_bounds__(128) void bbox_bias_kernel(
    const bf16_t* __restrict__ q,      // [B,S,NH*DH]
    const bf16_t* __restrict__ bbox,   // [B,S,S,DH]
    float* __restrict__ sc)            // [B,NH,S,S]
{
    const int wid  = threadIdx.x >> 5;
    const int lane = threadIdx.x & 31;
    const int i = blockIdx.x * 4 + wid;
    const int b = blockIdx.y;
    const int h  = lane & 15;
    const int hi = lane >> 4;

    const bf16_t* qrow = q + (long)(b * Sc + i) * Hc;
    v16bf fa[2];
    for (int kk = 0; kk < 2; kk++) {
        union { v16bf v; bf16_t e[16]; } u;
        for (int e2 = 0; e2 < 8; e2++) {
            int k1 = kk * 32 + hi * 8 + e2;
            int k2 = kk * 32 + 16 + hi * 8 + e2;
            u.e[e2]     = (h < NHc) ? qrow[h * DHc + k1] : (bf16_t)0.0f;
            u.e[8 + e2] = (h < NHc) ? qrow[h * DHc + k2] : (bf16_t)0.0f;
        }
        fa[kk] = u.v;
    }

    const bf16_t* bb = bbox + (long)(b * Sc + i) * Sc * DHc;
    for (int j0 = 0; j0 < Sc; j0 += 16) {
        v8f acc = {};
        int col = j0 + (lane & 15);
        for (int kk = 0; kk < 2; kk++) {
            int kb = kk * 32 + hi * 8;
            union { v16bf v; bf16x8 hh[2]; } u;
            const bf16_t* src = bb + (long)col * DHc + kb;
            u.hh[0] = *(const bf16x8*)src;
            u.hh[1] = *(const bf16x8*)(src + 16);
            acc = __builtin_amdgcn_wmma_f32_16x16x32_bf16(
                false, fa[kk], false, u.v, (short)0, acc, false, false);
        }
        for (int r = 0; r < 8; r++) {
            int hh = r + hi * 8;
            if (hh < NHc)
                sc[((((long)b * NHc + hh) * Sc + i) * Sc) + col] = acc[r];
        }
    }
}

// ---------------------------------------------------------------------------
// softmax over last dim (512), with *SCALE + mask fused. One block per row.
// ---------------------------------------------------------------------------
__global__ __launch_bounds__(256) void softmax_kernel(
    const float* __restrict__ sc, const float* __restrict__ mask, bf16_t* __restrict__ pr)
{
    int row = blockIdx.x;            // B*NH*S rows
    int b = row / (NHc * Sc);
    const float* p = sc + (long)row * Sc;
    int t = threadIdx.x;
    float x0 = p[t]       * SCALEc + mask[b * Sc + t];
    float x1 = p[t + 256] * SCALEc + mask[b * Sc + t + 256];

    __shared__ float redm[8], reds[8];
    int lane = t & 31, wid = t >> 5;

    float m = fmaxf(x0, x1);
    for (int o = 16; o; o >>= 1) m = fmaxf(m, __shfl_xor(m, o, 32));
    if (lane == 0) redm[wid] = m;
    __syncthreads();
    float mx = redm[0];
    for (int i = 1; i < 8; i++) mx = fmaxf(mx, redm[i]);

    float e0 = __expf(x0 - mx), e1 = __expf(x1 - mx);
    float s = e0 + e1;
    for (int o = 16; o; o >>= 1) s += __shfl_xor(s, o, 32);
    if (lane == 0) reds[wid] = s;
    __syncthreads();
    float tot = 0.f;
    for (int i = 0; i < 8; i++) tot += reds[i];
    float inv = 1.0f / tot;
    pr[(long)row * Sc + t]       = (bf16_t)(e0 * inv);
    pr[(long)row * Sc + t + 256] = (bf16_t)(e1 * inv);
}

// ---------------------------------------------------------------------------
// y = LayerNorm(a + res) * g + beta ; write f32 and bf16. One block per row.
// ---------------------------------------------------------------------------
__global__ __launch_bounds__(256) void add_ln_kernel(
    const float* __restrict__ a, const float* __restrict__ res,
    const float* __restrict__ g, const float* __restrict__ be,
    float* __restrict__ outF, bf16_t* __restrict__ outB)
{
    int row = blockIdx.x;
    int t = threadIdx.x;
    const float* pa = a + (long)row * Hc;
    const float* pr = res + (long)row * Hc;
    float v[3], s = 0.f, s2 = 0.f;
    for (int e = 0; e < 3; e++) {
        float x = pa[t + 256 * e] + pr[t + 256 * e];
        v[e] = x; s += x; s2 += x * x;
    }
    for (int o = 16; o; o >>= 1) { s += __shfl_xor(s, o, 32); s2 += __shfl_xor(s2, o, 32); }
    __shared__ float r1[8], r2[8];
    int lane = t & 31, wid = t >> 5;
    if (lane == 0) { r1[wid] = s; r2[wid] = s2; }
    __syncthreads();
    float ts = 0.f, ts2 = 0.f;
    for (int i = 0; i < 8; i++) { ts += r1[i]; ts2 += r2[i]; }
    float mu  = ts * (1.0f / Hc);
    float var = ts2 * (1.0f / Hc) - mu * mu;
    float inv = rsqrtf(var + 1e-12f);
    for (int e = 0; e < 3; e++) {
        int c = t + 256 * e;
        float y = (v[e] - mu) * inv * g[c] + be[c];
        if (outF) outF[(long)row * Hc + c] = y;
        if (outB) outB[(long)row * Hc + c] = (bf16_t)y;
    }
}

// ---------------------------------------------------------------------------
// Host: workspace carving + launch sequence.
// ---------------------------------------------------------------------------
extern "C" void kernel_launch(void* const* d_in, const int* in_sizes, int n_in,
                              void* d_out, int out_size, void* d_ws, size_t ws_size,
                              hipStream_t stream)
{
    (void)in_sizes; (void)n_in; (void)out_size; (void)ws_size;

    const float* hidden = (const float*)d_in[0];
    const float* bboxF  = (const float*)d_in[1];
    const float* maskp  = (const float*)d_in[2];
    const float* Wq = (const float*)d_in[3];  const float* bq  = (const float*)d_in[4];
    const float* Wk = (const float*)d_in[5];  const float* bk  = (const float*)d_in[6];
    const float* Wv = (const float*)d_in[7];  const float* bv  = (const float*)d_in[8];
    const float* Wo = (const float*)d_in[9];  const float* bo  = (const float*)d_in[10];
    const float* g1 = (const float*)d_in[11]; const float* be1 = (const float*)d_in[12];
    const float* Wi = (const float*)d_in[13]; const float* bi  = (const float*)d_in[14];
    const float* Wf = (const float*)d_in[15]; const float* bfb = (const float*)d_in[16];
    const float* g2 = (const float*)d_in[17]; const float* be2 = (const float*)d_in[18];

    const size_t WQE = (size_t)Lc * Hc * Hc;
    const size_t WIE = (size_t)Lc * Hc * Ic;
    const size_t BBE = (size_t)Bc * Sc * Sc * DHc;
    const size_t MH  = (size_t)Mrows * Hc;
    const size_t MI  = (size_t)Mrows * Ic;
    const size_t SCE = (size_t)Bc * NHc * Sc * Sc;

    char* p = (char*)d_ws;
    auto alloc = [&](size_t bytes) -> void* {
        void* r = (void*)p;
        p += (bytes + 255) & ~(size_t)255;
        return r;
    };
    bf16_t* wq16 = (bf16_t*)alloc(WQE * 2);
    bf16_t* wk16 = (bf16_t*)alloc(WQE * 2);
    bf16_t* wv16 = (bf16_t*)alloc(WQE * 2);
    bf16_t* wo16 = (bf16_t*)alloc(WQE * 2);
    bf16_t* wi16 = (bf16_t*)alloc(WIE * 2);
    bf16_t* wf16 = (bf16_t*)alloc(WIE * 2);
    bf16_t* bbox16 = (bf16_t*)alloc(BBE * 2);
    float*  h32  = (float*)alloc(MH * 4);
    float*  a32  = (float*)alloc(MH * 4);
    float*  t32  = (float*)alloc(MH * 4);
    bf16_t* hb   = (bf16_t*)alloc(MH * 2);
    bf16_t* ab   = (bf16_t*)alloc(MH * 2);
    bf16_t* q16  = (bf16_t*)alloc(MH * 2);
    bf16_t* k16  = (bf16_t*)alloc(MH * 2);
    bf16_t* v16  = (bf16_t*)alloc(MH * 2);
    bf16_t* ctx16 = (bf16_t*)alloc(MH * 2);
    bf16_t* ff16 = (bf16_t*)alloc(MI * 2);
    float*  sc32 = (float*)alloc(SCE * 4);
    bf16_t* pr16 = (bf16_t*)alloc(SCE * 2);

    auto cvt = [&](const float* in, bf16_t* out, size_t n) {
        cvt_kernel<<<(unsigned)((n + 255) / 256), 256, 0, stream>>>(in, out, (long)n);
    };
    // One-time (per call) precision conversion: weights fit in L2 as bf16.
    cvt(Wq, wq16, WQE); cvt(Wk, wk16, WQE); cvt(Wv, wv16, WQE); cvt(Wo, wo16, WQE);
    cvt(Wi, wi16, WIE); cvt(Wf, wf16, WIE);
    bbox_tr_kernel<<<(unsigned)((BBE + 255) / 256), 256, 0, stream>>>(bboxF, bbox16);
    (void)hipMemcpyAsync(h32, hidden, MH * sizeof(float), hipMemcpyDeviceToDevice, stream);
    cvt(hidden, hb, MH);

    const long SH = (long)Sc * Hc;       // per-batch stride of [B,S,H]
    const long SS = (long)Sc * Sc;       // per-(b,h) score stride

    for (int l = 0; l < Lc; l++) {
        const bf16_t* wq_l = wq16 + (size_t)l * Hc * Hc;
        const bf16_t* wk_l = wk16 + (size_t)l * Hc * Hc;
        const bf16_t* wv_l = wv16 + (size_t)l * Hc * Hc;
        const bf16_t* wo_l = wo16 + (size_t)l * Hc * Hc;
        const bf16_t* wi_l = wi16 + (size_t)l * Hc * Ic;
        const bf16_t* wf_l = wf16 + (size_t)l * Ic * Hc;

        // QKV projections: [2048,768] x [768,768] -> bf16 [B,S,NH,DH]
        gemm_kernel<false,false,false,false,true><<<dim3(6,16,1),256,0,stream>>>(
            hb, wq_l, bq + l*Hc, nullptr, q16, Mrows, Hc, Hc, Hc, Hc, Hc, 1,0,0,0,0,0,0);
        gemm_kernel<false,false,false,false,true><<<dim3(6,16,1),256,0,stream>>>(
            hb, wk_l, bk + l*Hc, nullptr, k16, Mrows, Hc, Hc, Hc, Hc, Hc, 1,0,0,0,0,0,0);
        gemm_kernel<false,false,false,false,true><<<dim3(6,16,1),256,0,stream>>>(
            hb, wv_l, bv + l*Hc, nullptr, v16, Mrows, Hc, Hc, Hc, Hc, Hc, 1,0,0,0,0,0,0);

        // bbox positional bias -> scores (writes entire tensor)
        bbox_bias_kernel<<<dim3(Sc/4, Bc), 128, 0, stream>>>(q16, bbox16, sc32);

        // content scores: q.k^T accumulated onto bbox bias; batched over z=b*12+h
        gemm_kernel<true,true,false,true,false><<<dim3(4,4,Bc*NHc),256,0,stream>>>(
            q16, k16, nullptr, sc32, nullptr, Sc, Sc, DHc, Hc, Hc, Sc,
            NHc, SH, 64L, SH, 64L, (long)NHc*SS, SS);

        // softmax (*SCALE + mask fused) -> bf16 probs
        softmax_kernel<<<Bc*NHc*Sc, 256, 0, stream>>>(sc32, maskp, pr16);

        // probs @ v -> ctx bf16 [B,S,NH,DH]
        gemm_kernel<false,false,false,false,true><<<dim3(1,4,Bc*NHc),256,0,stream>>>(
            pr16, v16, nullptr, nullptr, ctx16, Sc, DHc, Sc, Sc, Hc, Hc,
            NHc, (long)NHc*SS, SS, SH, 64L, SH, 64L);

        // attention output dense -> f32 (pre-LN)
        gemm_kernel<false,false,false,true,false><<<dim3(6,16,1),256,0,stream>>>(
            ctx16, wo_l, bo + l*Hc, t32, nullptr, Mrows, Hc, Hc, Hc, Hc, Hc, 1,0,0,0,0,0,0);

        // LN1: attn_out = LN(dense + x)
        add_ln_kernel<<<Mrows, 256, 0, stream>>>(t32, h32, g1 + l*Hc, be1 + l*Hc, a32, ab);

        // FFN up + exact GELU -> bf16
        gemm_kernel<false,false,true,false,true><<<dim3(24,16,1),256,0,stream>>>(
            ab, wi_l, bi + l*Ic, nullptr, ff16, Mrows, Ic, Hc, Hc, Ic, Ic, 1,0,0,0,0,0,0);

        // FFN down -> f32 (pre-LN)
        gemm_kernel<false,false,false,true,false><<<dim3(6,16,1),256,0,stream>>>(
            ff16, wf_l, bfb + l*Hc, t32, nullptr, Mrows, Hc, Ic, Ic, Hc, Hc, 1,0,0,0,0,0,0);

        // LN2: x = LN(ffn + attn_out); last layer writes d_out
        float* outF = (l == Lc - 1) ? (float*)d_out : h32;
        add_ln_kernel<<<Mrows, 256, 0, stream>>>(t32, a32, g2 + l*Hc, be2 + l*Hc, outF, hb);
    }
}
